// ViT_37039797960783
// MI455X (gfx1250) — compile-verified, matrix operand order checked by reading, not run on hardware
//
#include <hip/hip_runtime.h>

// ---------------------------------------------------------------------------
// ViT-B/16 forward for MI455X (gfx1250, wave32).  All GEMMs + attention run
// on v_wmma_f32_16x16x32_bf16 (bf16 inputs, fp32 accumulate).  GEMM tiles are
// staged with CDNA5 async global->LDS DMA (global_load_async_to_lds_b128,
// ASYNCcnt) into double-buffered LDS so tile k+1's DMA overlaps tile k's
// WMMAs.  All WMMA fragments are contiguous 16-byte loads (weights stored
// pre-transposed [N][K]; V transposed per-head to [64][224]).
// ---------------------------------------------------------------------------

#define USE_ASYNC_LDS 1

typedef __bf16 bf16;
typedef __attribute__((ext_vector_type(16))) bf16  v16bf;
typedef __attribute__((ext_vector_type(8)))  bf16  v8bf;
typedef __attribute__((ext_vector_type(8)))  float v8f;

constexpr int kD  = 768;
constexpr int kH  = 12;
constexpr int kDH = 64;
constexpr int kS  = 197;
constexpr int kSP = 224;         // padded sequence (7 * 32) for P*V K-dim
constexpr int kB  = 32;
constexpr int kBS = kB * kS;     // 6304
constexpr int kNP = 196;
constexpr int kPB = kB * kNP;    // 6272
constexpr int kDM = 3072;
constexpr int kNC = 82;

static inline unsigned cdiv(size_t a, size_t b) { return (unsigned)((a + b - 1) / b); }

// Build a 16x32 (or 32x16) WMMA fragment for this lane from two contiguous
// 8-element (16B) chunks, per the ISA layout: elems 0-7 at K = hi*8,
// elems 8-15 at K = 16 + hi*8, where hi = lane>>4.
static __device__ __forceinline__ v16bf load_frag(const bf16* __restrict__ base, int hi) {
  const v8bf lo  = *reinterpret_cast<const v8bf*>(base + hi * 8);
  const v8bf hi8 = *reinterpret_cast<const v8bf*>(base + 16 + hi * 8);
  return __builtin_shufflevector(lo, hi8, 0, 1, 2, 3, 4, 5, 6, 7,
                                 8, 9, 10, 11, 12, 13, 14, 15);
}

#if USE_ASYNC_LDS
// LDS byte offset of a __shared__ object (addrspacecast -> ptrtoint)
static __device__ __forceinline__ uint32_t lds_addr_of(const void* p) {
  return (uint32_t)(uintptr_t)(__attribute__((address_space(3))) const void*)p;
}
// 32B global->LDS async copy (two b128 lanes ops; offset applies to both sides)
static __device__ __forceinline__ void async_copy32(uint32_t lds, const bf16* g) {
  asm volatile("global_load_async_to_lds_b128 %0, %1, off\n\t"
               "global_load_async_to_lds_b128 %0, %1, off offset:16"
               :: "v"(lds), "v"(g) : "memory");
}
#endif

// ---------------------------------------------------------------------------
// Generic bf16 WMMA GEMM:  C[M,N] = A[M,K] * Bt[N,K]^T  (+bias, gelu, resid)
// Bt is the weight matrix stored PRE-TRANSPOSED as [N][K] row-major.
// Block tile 128x128, K-step 32, double-buffered LDS.  8 waves, 32x64 each.
// N is always a multiple of 128 for every call site; M is guarded.
// ---------------------------------------------------------------------------
__global__ __launch_bounds__(256) void k_gemm(
    const bf16* __restrict__ A, const bf16* __restrict__ Bt,
    const float* __restrict__ bias, const float* __restrict__ resid,
    float* __restrict__ outF, bf16* __restrict__ outB,
    int M, int N, int K, int act)
{
  __shared__ bf16 As[2][128][40];   // row stride 80B (16B aligned)
  __shared__ bf16 Bs[2][128][40];   // [col][k]

  const int tid  = threadIdx.x;
  const int lane = tid & 31;
  const int w    = tid >> 5;
  const int wm   = w & 3;        // 4 waves over M (32 rows each)
  const int wn   = w >> 2;       // 2 waves over N (64 cols each)
  const int bm   = blockIdx.x * 128;
  const int bn   = blockIdx.y * 128;
  const int nl   = lane & 15;
  const int hi   = lane >> 4;

  v8f acc[2][4];
#pragma unroll
  for (int i = 0; i < 2; ++i)
#pragma unroll
    for (int j = 0; j < 4; ++j)
#pragma unroll
      for (int r = 0; r < 8; ++r) acc[i][j][r] = 0.f;

  const int sr = tid >> 1, sh = tid & 1;   // staging: row/col, 16-elem half
  const int gmc = (bm + sr < M) ? (bm + sr) : (M - 1);   // clamped A row
  const int gnc = bn + sr;
  const bf16* aRow = A  + (size_t)gmc * K + sh * 16;
  const bf16* bRow = Bt + (size_t)gnc * K + sh * 16;

#if USE_ASYNC_LDS
  const uint32_t ldsA0 = lds_addr_of(&As[0][sr][sh * 16]);
  const uint32_t ldsA1 = lds_addr_of(&As[1][sr][sh * 16]);
  const uint32_t ldsB0 = lds_addr_of(&Bs[0][sr][sh * 16]);
  const uint32_t ldsB1 = lds_addr_of(&Bs[1][sr][sh * 16]);
  // prologue: stage tile 0 into buffer 0
  async_copy32(ldsA0, aRow);
  async_copy32(ldsB0, bRow);
#endif

  int buf = 0;
  for (int k0 = 0; k0 < K; k0 += 32) {
#if USE_ASYNC_LDS
    if (k0 + 32 < K) {   // kick DMA for the next tile into the other buffer
      async_copy32(buf ? ldsA0 : ldsA1, aRow + k0 + 32);
      async_copy32(buf ? ldsB0 : ldsB1, bRow + k0 + 32);
      __builtin_prefetch(aRow + k0 + 64, 0, 2);   // speculative, L2 -> WGP$
      __builtin_prefetch(bRow + k0 + 64, 0, 2);
      // async loads complete in order: <=4 outstanding means tile k0 landed
      asm volatile("s_wait_asynccnt 0x4" ::: "memory");
    } else {
      asm volatile("s_wait_asynccnt 0x0" ::: "memory");
    }
#else
    {
      const uint4* sa = reinterpret_cast<const uint4*>(aRow + k0);
      const uint4* sb = reinterpret_cast<const uint4*>(bRow + k0);
      const uint4 a0 = sa[0], a1 = sa[1], b0 = sb[0], b1 = sb[1];
      uint4* da = reinterpret_cast<uint4*>(&As[buf][sr][sh * 16]);
      uint4* db = reinterpret_cast<uint4*>(&Bs[buf][sr][sh * 16]);
      da[0] = a0; da[1] = a1; db[0] = b0; db[1] = b1;
    }
#endif
    __syncthreads();   // tile data visible to all waves

    v16bf af[2];
#pragma unroll
    for (int i = 0; i < 2; ++i)
      af[i] = load_frag(&As[buf][wm * 32 + i * 16 + nl][0], hi);
    v16bf bf[4];
#pragma unroll
    for (int j = 0; j < 4; ++j)
      bf[j] = load_frag(&Bs[buf][wn * 64 + j * 16 + nl][0], hi);

#pragma unroll
    for (int i = 0; i < 2; ++i)
#pragma unroll
      for (int j = 0; j < 4; ++j)
        acc[i][j] = __builtin_amdgcn_wmma_f32_16x16x32_bf16(
            false, af[i], false, bf[j], (short)0, acc[i][j], false, false);
    __syncthreads();   // WAR: everyone done reading before buffer is refilled
    buf ^= 1;
  }

  // epilogue: D layout -> m = r + 8*(lane>>4), n = lane&15
#pragma unroll
  for (int j = 0; j < 4; ++j) {
    const int n = bn + wn * 64 + j * 16 + nl;
    const float bv = bias ? bias[n] : 0.f;
#pragma unroll
    for (int i = 0; i < 2; ++i) {
#pragma unroll
      for (int r = 0; r < 8; ++r) {
        const int m = bm + wm * 32 + i * 16 + r + hi * 8;
        if (m >= M) continue;
        float v = acc[i][j][r] + bv;
        if (act == 1) v = 0.5f * v * (1.f + erff(v * 0.70710678118f));
        const size_t idx = (size_t)m * N + n;
        if (resid) v += resid[idx];
        if (outF) outF[idx] = v;
        if (outB) outB[idx] = (bf16)v;
      }
    }
  }
}

// ---------------------------------------------------------------------------
// Fused attention: per (b, head, 16-row tile).  Scores/softmax/P*V via WMMA.
// Q/K stored as [B*S, 768] (head h in cols [h*64, h*64+64)).
// Vt stored transposed per (b,h): [64][224] with zero-padded tail.
// Address clamping replaces per-element masks: masked score columns get -inf
// before softmax, and padded P columns are exactly zero, so clamped loads
// never contaminate stored results.
// ---------------------------------------------------------------------------
__global__ __launch_bounds__(128) void k_attn(
    const bf16* __restrict__ Qb, const bf16* __restrict__ Kb,
    const bf16* __restrict__ Vt, bf16* __restrict__ Cb)
{
  __shared__ bf16 Ps[4][16][232];  // per-wave probability tile (16B-mult stride)

  const int lane = threadIdx.x & 31;
  const int w    = threadIdx.x >> 5;
  const int nl   = lane & 15;
  const int hi   = lane >> 4;
  const int b    = blockIdx.z;
  const int hh   = blockIdx.y;
  int mt = blockIdx.x * 4 + w;
  const bool valid = (mt < 13);
  if (mt > 12) mt = 12;
  const int s0 = mt * 16;

  const size_t base = ((size_t)b * kS) * kD + (size_t)hh * kDH;
  const bf16* Qp  = Qb + base;
  const bf16* Kp  = Kb + base;
  bf16*       Cp  = Cb + base;
  const bf16* Vtp = Vt + (size_t)(b * kH + hh) * kDH * kSP;

  // Q fragments (16x64 = two 16x32 A fragments), clamped row address
  v16bf qf[2];
  {
    const int rowc = s0 + nl < kS ? s0 + nl : kS - 1;
    const bf16* p = Qp + (size_t)rowc * kD;
#pragma unroll
    for (int ks = 0; ks < 2; ++ks) qf[ks] = load_frag(p + ks * 32, hi);
  }

  // scores = Q K^T * scale  (13 column tiles of 16)
  float sc[13][8];
  const float scale = 0.125f;  // 1/sqrt(64)
#pragma unroll
  for (int j = 0; j < 13; ++j) {
    const int  t  = j * 16 + nl;
    const bool ok = t < kS;
    const bf16* kp = Kp + (size_t)(ok ? t : kS - 1) * kD;
    v8f a;
#pragma unroll
    for (int r = 0; r < 8; ++r) a[r] = 0.f;
#pragma unroll
    for (int ks = 0; ks < 2; ++ks) {
      const v16bf kb = load_frag(kp + ks * 32, hi);
      a = __builtin_amdgcn_wmma_f32_16x16x32_bf16(
          false, qf[ks], false, kb, (short)0, a, false, false);
    }
#pragma unroll
    for (int r = 0; r < 8; ++r) sc[j][r] = ok ? a[r] * scale : -1e30f;
  }

  // row softmax: each row's 16 cols per tile live across one 16-lane half
#pragma unroll
  for (int r = 0; r < 8; ++r) {
    float mx = -1e30f;
#pragma unroll
    for (int j = 0; j < 13; ++j) mx = fmaxf(mx, sc[j][r]);
#pragma unroll
    for (int m = 8; m >= 1; m >>= 1) mx = fmaxf(mx, __shfl_xor(mx, m, 32));
    float sum = 0.f;
#pragma unroll
    for (int j = 0; j < 13; ++j) {
      const float e = __expf(sc[j][r] - mx);
      sc[j][r] = e;
      sum += e;
    }
#pragma unroll
    for (int m = 8; m >= 1; m >>= 1) sum += __shfl_xor(sum, m, 32);
    const float inv = 1.f / sum;
#pragma unroll
    for (int j = 0; j < 13; ++j) sc[j][r] *= inv;
  }

  // spill probabilities to LDS (C layout -> A layout re-fragmentation)
#pragma unroll
  for (int j = 0; j < 13; ++j)
#pragma unroll
    for (int r = 0; r < 8; ++r)
      Ps[w][r + hi * 8][j * 16 + nl] = (bf16)sc[j][r];
  for (int idx = lane; idx < 16 * 16; idx += 32)      // zero cols [208,224)
    Ps[w][idx >> 4][208 + (idx & 15)] = (bf16)0.f;

  // ctx = P[16x224] * V[224x64]; V fragments contiguous thanks to Vt layout
  v8f cacc[4];
#pragma unroll
  for (int sn = 0; sn < 4; ++sn)
#pragma unroll
    for (int r = 0; r < 8; ++r) cacc[sn][r] = 0.f;

#pragma unroll
  for (int j = 0; j < 7; ++j) {
    const v16bf af = load_frag(&Ps[w][nl][j * 32], hi);
#pragma unroll
    for (int sn = 0; sn < 4; ++sn) {
      const v16bf vb = load_frag(Vtp + (size_t)(sn * 16 + nl) * kSP + j * 32, hi);
      cacc[sn] = __builtin_amdgcn_wmma_f32_16x16x32_bf16(
          false, af, false, vb, (short)0, cacc[sn], false, false);
    }
  }

#pragma unroll
  for (int sn = 0; sn < 4; ++sn)
#pragma unroll
    for (int r = 0; r < 8; ++r) {
      const int row = s0 + r + hi * 8;
      if (valid && row < kS)
        Cp[(size_t)row * kD + sn * 16 + nl] = (bf16)cacc[sn][r];
    }
}

// ---------------------------------------------------------------------------
// LayerNorm (fp32 in, bf16 out), one row (768) per block of 256 threads
// ---------------------------------------------------------------------------
static __device__ __forceinline__ float blk_reduce(float v, float* sm) {
#pragma unroll
  for (int m = 16; m >= 1; m >>= 1) v += __shfl_xor(v, m, 32);
  const int lane = threadIdx.x & 31, w = threadIdx.x >> 5;
  __syncthreads();
  if (lane == 0) sm[w] = v;
  __syncthreads();
  float r = sm[0];
#pragma unroll
  for (int i = 1; i < 8; ++i) r += sm[i];
  return r;
}

__global__ __launch_bounds__(256) void k_ln(
    const float* __restrict__ h, const float* __restrict__ g,
    const float* __restrict__ bta, bf16* __restrict__ out)
{
  __shared__ float sm[8];
  const int row = blockIdx.x;
  const float* x = h + (size_t)row * kD;
  float s = 0.f;
  for (int i = threadIdx.x; i < kD; i += 256) s += x[i];
  const float mean = blk_reduce(s, sm) * (1.f / kD);
  float v = 0.f;
  for (int i = threadIdx.x; i < kD; i += 256) {
    const float d = x[i] - mean;
    v += d * d;
  }
  const float var = blk_reduce(v, sm) * (1.f / kD);
  const float inv = rsqrtf(var + 1e-5f);
  bf16* o = out + (size_t)row * kD;
  for (int i = threadIdx.x; i < kD; i += 256)
    o[i] = (bf16)((x[i] - mean) * inv * g[i] + bta[i]);
}

// ---------------------------------------------------------------------------
// Data-prep / small kernels
// ---------------------------------------------------------------------------
__global__ void k_f2b(const float* __restrict__ s, bf16* __restrict__ d, size_t n) {
  const size_t i = (size_t)blockIdx.x * blockDim.x + threadIdx.x;
  if (i < n) d[i] = (bf16)s[i];
}

// transpose-convert: src [K][N] fp32 -> dst [N][K] bf16
__global__ void k_f2bT(const float* __restrict__ s, bf16* __restrict__ d, int K, int N) {
  const size_t i = (size_t)blockIdx.x * blockDim.x + threadIdx.x;
  if (i >= (size_t)K * N) return;
  const int n = (int)(i / K);
  const int k = (int)(i % K);
  d[i] = (bf16)s[(size_t)k * N + n];
}

// Wq/Wk/Wv [H, D, DH] -> bf16 Bt[N=768][K=768]; n = h*64+e, k = d
__global__ void k_qkvw(const float* __restrict__ W, bf16* __restrict__ dst) {
  const size_t i = (size_t)blockIdx.x * blockDim.x + threadIdx.x;
  if (i >= (size_t)kD * kD) return;
  const int n = (int)(i / kD);
  const int d = (int)(i % kD);
  const int hq = n >> 6, e = n & 63;
  dst[i] = (bf16)W[((size_t)hq * kD + d) * kDH + e];
}

// V [B*S, 768] -> Vt [B][H][64][224] (zero-padded tail rows)
__global__ void k_vT(const bf16* __restrict__ V, bf16* __restrict__ Vt) {
  const size_t i = (size_t)blockIdx.x * blockDim.x + threadIdx.x;
  if (i >= (size_t)kB * kH * kDH * kSP) return;
  const int t = (int)(i % kSP);
  const size_t r = i / kSP;
  const int e  = (int)(r % kDH);
  const size_t r2 = r / kDH;
  const int hh = (int)(r2 % kH);
  const int b  = (int)(r2 / kH);
  Vt[i] = (t < kS) ? V[((size_t)b * kS + t) * kD + hh * kDH + e] : (bf16)0.f;
}

// x [B,3,224,224] -> patches bf16 [B*196, 768]  (cij ordering)
__global__ void k_im2col(const float* __restrict__ x, bf16* __restrict__ dst) {
  const size_t i = (size_t)blockIdx.x * blockDim.x + threadIdx.x;
  if (i >= (size_t)kPB * kD) return;
  const int q = (int)(i % kD);
  const size_t rp = i / kD;
  const int p  = (int)(rp % kNP);
  const int bb = (int)(rp / kNP);
  const int c  = q >> 8;
  const int ii = (q & 255) >> 4;
  const int jj = q & 15;
  const int pr = p / 14, pc = p % 14;
  const size_t src = (((size_t)bb * 3 + c) * 224 + (pr * 16 + ii)) * 224 + (pc * 16 + jj);
  dst[i] = (bf16)x[src];
}

// h[b,s,d] = (s==0 ? cls[d] : hp[b,s-1,d]) + pos[s,d]
__global__ void k_clspos(const float* __restrict__ hp, const float* __restrict__ cls,
                         const float* __restrict__ pos, float* __restrict__ h) {
  const size_t i = (size_t)blockIdx.x * blockDim.x + threadIdx.x;
  if (i >= (size_t)kBS * kD) return;
  const int d = (int)(i % kD);
  const size_t r = i / kD;
  const int s = (int)(r % kS);
  const int b = (int)(r / kS);
  const float v = (s == 0) ? cls[d] : hp[((size_t)b * kNP + (s - 1)) * kD + d];
  h[i] = v + pos[(size_t)s * kD + d];
}

// head: t[b,d] = tanh(h[b,0,:] . Wh[:,d] + bh[d]);  out[b,n] = t[b,:] . Wc[:,n] + bc[n]
__global__ void k_head1(const float* __restrict__ h, const float* __restrict__ Wh,
                        const float* __restrict__ bh, float* __restrict__ t) {
  const int b = blockIdx.x;
  const float* x = h + (size_t)b * kS * kD;  // row s=0
  for (int d = threadIdx.x; d < kD; d += 256) {
    float s = bh[d];
    for (int k = 0; k < kD; ++k) s += x[k] * Wh[(size_t)k * kD + d];
    t[(size_t)b * kD + d] = tanhf(s);
  }
}

__global__ void k_head2(const float* __restrict__ t, const float* __restrict__ Wc,
                        const float* __restrict__ bc, float* __restrict__ out) {
  const int b = blockIdx.x;
  for (int n = threadIdx.x; n < kNC; n += 256) {
    float s = bc[n];
    for (int k = 0; k < kD; ++k) s += t[(size_t)b * kD + k] * Wc[(size_t)k * kNC + n];
    out[(size_t)b * kNC + n] = s;
  }
}

// ---------------------------------------------------------------------------
// Host orchestration
// ---------------------------------------------------------------------------
extern "C" void kernel_launch(void* const* d_in, const int* in_sizes, int n_in,
                              void* d_out, int out_size, void* d_ws, size_t ws_size,
                              hipStream_t stream) {
  const float* x      = (const float*)d_in[0];
  const float* conv_w = (const float*)d_in[1];
  const float* conv_b = (const float*)d_in[2];
  const float* cls    = (const float*)d_in[3];
  const float* pos    = (const float*)d_in[4];
  const float* Wq = (const float*)d_in[5];
  const float* bq = (const float*)d_in[6];
  const float* Wk = (const float*)d_in[7];
  const float* bk = (const float*)d_in[8];
  const float* Wv = (const float*)d_in[9];
  const float* bv = (const float*)d_in[10];
  const float* Wo = (const float*)d_in[11];
  const float* bo = (const float*)d_in[12];
  const float* ln1w = (const float*)d_in[13];
  const float* ln1b = (const float*)d_in[14];
  const float* W1 = (const float*)d_in[15];
  const float* b1 = (const float*)d_in[16];
  const float* W2 = (const float*)d_in[17];
  const float* b2 = (const float*)d_in[18];
  const float* Wh = (const float*)d_in[19];
  const float* bh = (const float*)d_in[20];
  const float* Wc = (const float*)d_in[21];
  const float* bc = (const float*)d_in[22];

  // workspace carve-up
  char* cur = (char*)d_ws;
  auto take = [&](size_t bytes) {
    void* p = cur;
    cur += (bytes + 255) & ~(size_t)255;
    return p;
  };
  float* h    = (float*)take((size_t)kBS * kD * 4);
  float* hp   = (float*)take((size_t)kPB * kD * 4);
  float* thd  = (float*)take((size_t)kB * kD * 4);
  bf16* a_in  = (bf16*)take((size_t)kBS * kD * 2);
  bf16* hb    = (bf16*)take((size_t)kBS * kD * 2);
  bf16* Qb    = (bf16*)take((size_t)kBS * kD * 2);
  bf16* Kb    = (bf16*)take((size_t)kBS * kD * 2);
  bf16* Vb    = (bf16*)take((size_t)kBS * kD * 2);
  bf16* Vt    = (bf16*)take((size_t)kB * kH * kDH * kSP * 2);
  bf16* ctxb  = (bf16*)take((size_t)kBS * kD * 2);
  bf16* m1b   = (bf16*)take((size_t)kBS * kDM * 2);
  bf16* wscr  = (bf16*)take((size_t)kD * kDM * 2);
  bf16* Xp    = (bf16*)take((size_t)kPB * kD * 2);
  bf16* cwT   = (bf16*)take((size_t)kD * kD * 2);

  const dim3 blk256(256);

  // patch embedding as GEMM (conv_w [D][cij] is already the [N][K] form)
  k_im2col<<<cdiv((size_t)kPB * kD, 256), blk256, 0, stream>>>(x, Xp);
  k_f2b<<<cdiv((size_t)kD * kD, 256), blk256, 0, stream>>>(conv_w, cwT, (size_t)kD * kD);
  k_gemm<<<dim3(cdiv(kPB, 128), cdiv(kD, 128)), blk256, 0, stream>>>(
      Xp, cwT, conv_b, nullptr, hp, nullptr, kPB, kD, kD, 0);
  k_clspos<<<cdiv((size_t)kBS * kD, 256), blk256, 0, stream>>>(hp, cls, pos, h);

  const dim3 gD(cdiv(kBS, 128), cdiv(kD, 128));
  const dim3 gDM(cdiv(kBS, 128), cdiv(kDM, 128));

  for (int l = 0; l < 12; ++l) {
    const size_t wl  = (size_t)l * kD * kD;       // per-layer D x D weights
    const size_t wml = (size_t)l * kD * kDM;      // per-layer D x DM weights

    k_ln<<<kBS, blk256, 0, stream>>>(h, ln1w + (size_t)l * kD, ln1b + (size_t)l * kD, a_in);

    // Q, K, V projections (weights emitted as Bt [N][K])
    k_qkvw<<<cdiv((size_t)kD * kD, 256), blk256, 0, stream>>>(Wq + wl, wscr);
    k_gemm<<<gD, blk256, 0, stream>>>(a_in, wscr, bq + (size_t)l * kD, nullptr,
                                      nullptr, Qb, kBS, kD, kD, 0);
    k_qkvw<<<cdiv((size_t)kD * kD, 256), blk256, 0, stream>>>(Wk + wl, wscr);
    k_gemm<<<gD, blk256, 0, stream>>>(a_in, wscr, bk + (size_t)l * kD, nullptr,
                                      nullptr, Kb, kBS, kD, kD, 0);
    k_qkvw<<<cdiv((size_t)kD * kD, 256), blk256, 0, stream>>>(Wv + wl, wscr);
    k_gemm<<<gD, blk256, 0, stream>>>(a_in, wscr, bv + (size_t)l * kD, nullptr,
                                      nullptr, Vb, kBS, kD, kD, 0);
    k_vT<<<cdiv((size_t)kB * kH * kDH * kSP, 256), blk256, 0, stream>>>(Vb, Vt);

    // attention (13 row tiles, 12 heads, 32 batches)
    k_attn<<<dim3(4, kH, kB), dim3(128), 0, stream>>>(Qb, Kb, Vt, ctxb);

    // output projection + residual (writes fp32 h in place, bf16 copy hb)
    k_f2bT<<<cdiv((size_t)kD * kD, 256), blk256, 0, stream>>>(Wo + wl, wscr, kD, kD);
    k_gemm<<<gD, blk256, 0, stream>>>(ctxb, wscr, bo + (size_t)l * kD, h,
                                      h, hb, kBS, kD, kD, 0);

    // MLP (takes un-normalized residual stream, matching reference)
    k_f2bT<<<cdiv((size_t)kD * kDM, 256), blk256, 0, stream>>>(W1 + wml, wscr, kD, kDM);
    k_gemm<<<gDM, blk256, 0, stream>>>(hb, wscr, b1 + (size_t)l * kDM, nullptr,
                                       nullptr, m1b, kBS, kDM, kD, 1 /*gelu*/);
    k_f2bT<<<cdiv((size_t)kDM * kD, 256), blk256, 0, stream>>>(W2 + wml, wscr, kDM, kD);
    k_gemm<<<gD, blk256, 0, stream>>>(m1b, wscr, b2 + (size_t)l * kD, h,
                                      h, nullptr, kBS, kD, kDM, 0);
  }

  // classifier head (tiny; plain fp32)
  k_head1<<<kB, blk256, 0, stream>>>(h, Wh, bh, thd);
  k_head2<<<kB, blk256, 0, stream>>>(thd, Wc, bc, (float*)d_out);
}
